// LSConv_1013612282170
// MI455X (gfx1250) — matrix operand order repaired and checked by reading, not compiled, not run on hardware
//
#include <hip/hip_runtime.h>
#include <hip/hip_bf16.h>

typedef __attribute__((ext_vector_type(16))) __bf16 v16bf;
typedef __attribute__((ext_vector_type(8)))  __bf16 v8bf;
typedef __attribute__((ext_vector_type(8)))  float  v8f;

#define DIMC  256
#define MIDC  128
#define WCHC  288
#define HD    80
#define WD    80
#define SP    6400
#define NB    16
#define EPSF  1e-5f

__device__ __forceinline__ int kmap_e(int e) { return e + ((e >= 8) ? 8 : 0); }

// ---------------- prep: fold BN into scales/biases, convert weights to bf16 ----
__global__ __launch_bounds__(256)
void k_prep(const float* __restrict__ cv1_w, const float* __restrict__ cv1_g,
            const float* __restrict__ cv1_b, const float* __restrict__ cv1_m,
            const float* __restrict__ cv1_v,
            const float* __restrict__ cv2_w, const float* __restrict__ cv2_g,
            const float* __restrict__ cv2_b, const float* __restrict__ cv2_m,
            const float* __restrict__ cv2_v,
            const float* __restrict__ cv3_w, const float* __restrict__ cv3_g,
            const float* __restrict__ cv3_b, const float* __restrict__ cv3_m,
            const float* __restrict__ cv3_v,
            const float* __restrict__ cv4_w,
            const float* __restrict__ bn_g, const float* __restrict__ bn_b,
            const float* __restrict__ bn_m, const float* __restrict__ bn_v,
            __bf16* __restrict__ w1, __bf16* __restrict__ w3,
            __bf16* __restrict__ w4, float* __restrict__ wdw,
            float* __restrict__ prm)
{
    int i = blockIdx.x * 256 + threadIdx.x;   // grid covers 288*128 = 36864
    if (i < 128 * 256) w1[i] = (__bf16)cv1_w[i];
    if (i < 128 * 128) w3[i] = (__bf16)cv3_w[i];
    w4[i] = (__bf16)cv4_w[i];
    if (i < 128) {
        float s1 = cv1_g[i] * rsqrtf(cv1_v[i] + EPSF);
        prm[0   + i] = s1;
        prm[128 + i] = cv1_b[i] - cv1_m[i] * s1;
        float s2 = cv2_g[i] * rsqrtf(cv2_v[i] + EPSF);
        prm[256 + i] = cv2_b[i] - cv2_m[i] * s2;
        for (int t = 0; t < 49; ++t) wdw[i * 49 + t] = cv2_w[i * 49 + t] * s2;
        float s3 = cv3_g[i] * rsqrtf(cv3_v[i] + EPSF);
        prm[384 + i] = s3;
        prm[512 + i] = cv3_b[i] - cv3_m[i] * s3;
    }
    if (i < 256) {
        float bs = bn_g[i] * rsqrtf(bn_v[i] + EPSF);
        prm[640 + i] = bs;
        prm[896 + i] = bn_b[i] - bn_m[i] * bs;
    }
}

// ---------------- cv1: 1x1 conv 256->128 GEMM + BN + ReLU (WMMA bf16) --------
__global__ __launch_bounds__(256)
void k_cv1(const float* __restrict__ x, const __bf16* __restrict__ w1,
           const float* __restrict__ prm, __bf16* __restrict__ h1)
{
    int lane = threadIdx.x & 31;
    int wave = threadIdx.x >> 5;
    int T    = blockIdx.x * 8 + wave;       // 0..6399 pixel tiles
    int b    = T / 400;
    int sp0  = (T % 400) * 16;
    int m    = lane & 15;
    int half = lane >> 4;
    const float* xb = x + (size_t)b * DIMC * SP;

    v8f acc[8];
#pragma unroll
    for (int i = 0; i < 8; ++i) acc[i] = (v8f){0, 0, 0, 0, 0, 0, 0, 0};

#pragma unroll
    for (int ks = 0; ks < 8; ++ks) {
        int k0 = ks * 32;
        v16bf a;
#pragma unroll
        for (int e = 0; e < 16; ++e) {
            int ch = k0 + e + 16 * half;
            a[e] = (__bf16)xb[(size_t)ch * SP + sp0 + m];
        }
#pragma unroll
        for (int nt = 0; nt < 8; ++nt) {
            int n = nt * 16 + m;
            v16bf bm = *(const v16bf*)(w1 + (size_t)n * DIMC + k0 + 16 * half);
            acc[nt] = __builtin_amdgcn_wmma_f32_16x16x32_bf16(
                false, a, false, bm, (short)0, acc[nt], false, false);
        }
    }
#pragma unroll
    for (int nt = 0; nt < 8; ++nt) {
        int n = nt * 16 + m;
        float s = prm[0 + n];
        float t = prm[128 + n];
        v8bf pk;
#pragma unroll
        for (int r = 0; r < 8; ++r)
            pk[r] = (__bf16)fmaxf(acc[nt][r] * s + t, 0.f);
        *(v8bf*)(h1 + ((size_t)(b * MIDC + n)) * SP + sp0 + 8 * half) = pk;
    }
}

// ---------------- cv2: depthwise 7x7 + folded BN ------------------------------
__global__ __launch_bounds__(256)
void k_dw7(const __bf16* __restrict__ h1, const float* __restrict__ wdw,
           const float* __restrict__ prm, __bf16* __restrict__ h2)
{
    int idx = blockIdx.x * 256 + threadIdx.x;   // 16*128*6400
    int sp  = idx % SP;
    int bc  = idx / SP;
    int c   = bc % MIDC;
    int h = sp / WD, w = sp % WD;
    const __bf16* src = h1 + (size_t)bc * SP;
    const float*  wt  = wdw + c * 49;
    float acc = prm[256 + c];
#pragma unroll
    for (int dy = -3; dy <= 3; ++dy) {
        int hh = h + dy;
        if ((unsigned)hh < HD) {
#pragma unroll
            for (int dx = -3; dx <= 3; ++dx) {
                int ww = w + dx;
                if ((unsigned)ww < WD)
                    acc += (float)src[hh * WD + ww] * wt[(dy + 3) * 7 + (dx + 3)];
            }
        }
    }
    h2[idx] = (__bf16)acc;
}

// ---------------- cv3 + ReLU fused with cv4 (+bias) via LDS transpose ---------
__global__ __launch_bounds__(256)
void k_cv34(const __bf16* __restrict__ h2, const __bf16* __restrict__ w3,
            const __bf16* __restrict__ w4, const float* __restrict__ cv4_b,
            const float* __restrict__ prm, __bf16* __restrict__ wraw)
{
    __shared__ __bf16 tbuf[8 * 128 * 16];   // per-wave [ch][pix] tile, 32KB
    int lane = threadIdx.x & 31;
    int wave = threadIdx.x >> 5;
    int T    = blockIdx.x * 8 + wave;
    int b    = T / 400;
    int sp0  = (T % 400) * 16;
    int m    = lane & 15;
    int half = lane >> 4;
    const __bf16* hb = h2 + (size_t)b * MIDC * SP;
    int wbase = wave * 128 * 16;

    // GEMM1: t = relu(BN(h2 @ W3^T)), K = 128
    v8f acc[8];
#pragma unroll
    for (int i = 0; i < 8; ++i) acc[i] = (v8f){0, 0, 0, 0, 0, 0, 0, 0};
#pragma unroll
    for (int ks = 0; ks < 4; ++ks) {
        int k0 = ks * 32;
        v16bf a;
#pragma unroll
        for (int e = 0; e < 16; ++e)
            a[e] = hb[(size_t)(k0 + e + 16 * half) * SP + sp0 + m];
#pragma unroll
        for (int nt = 0; nt < 8; ++nt) {
            int n = nt * 16 + m;
            v16bf bm = *(const v16bf*)(w3 + (size_t)n * MIDC + k0 + 16 * half);
            acc[nt] = __builtin_amdgcn_wmma_f32_16x16x32_bf16(
                false, a, false, bm, (short)0, acc[nt], false, false);
        }
    }
    // write tile to LDS (packed pairs of consecutive pixels)
    unsigned* tbu = (unsigned*)tbuf;
#pragma unroll
    for (int nt = 0; nt < 8; ++nt) {
        int n = nt * 16 + m;
        float s = prm[384 + n], t = prm[512 + n];
#pragma unroll
        for (int r = 0; r < 8; r += 2) {
            union { __bf16 hv[2]; unsigned u; } pk;
            pk.hv[0] = (__bf16)fmaxf(acc[nt][r]     * s + t, 0.f);
            pk.hv[1] = (__bf16)fmaxf(acc[nt][r + 1] * s + t, 0.f);
            tbu[(wbase + n * 16 + 8 * half + r) >> 1] = pk.u;
        }
    }
    asm volatile("s_wait_dscnt 0x0" ::: "memory");

    // re-read in A layout (transpose through LDS)
    v16bf a2[4];
#pragma unroll
    for (int ks = 0; ks < 4; ++ks) {
#pragma unroll
        for (int e = 0; e < 16; ++e)
            a2[ks][e] = tbuf[wbase + (ks * 32 + e + 16 * half) * 16 + m];
    }

    // GEMM2: wraw = t @ W4^T + bias, N = 288 (18 tiles)
    __bf16* wrb = wraw + (size_t)b * WCHC * SP;
#pragma unroll
    for (int nt = 0; nt < 18; ++nt) {
        int n = nt * 16 + m;
        v8f c2 = (v8f){0, 0, 0, 0, 0, 0, 0, 0};
#pragma unroll
        for (int ks = 0; ks < 4; ++ks) {
            v16bf bm = *(const v16bf*)(w4 + (size_t)n * MIDC + ks * 32 + 16 * half);
            c2 = __builtin_amdgcn_wmma_f32_16x16x32_bf16(
                false, a2[ks], false, bm, (short)0, c2, false, false);
        }
        float bias = cv4_b[n];
        v8bf pk;
#pragma unroll
        for (int r = 0; r < 8; ++r) pk[r] = (__bf16)(c2[r] + bias);
        *(v8bf*)(wrb + (size_t)n * SP + sp0 + 8 * half) = pk;
    }
}

// ---------------- GroupNorm statistics (32 groups x 9ch, per batch) -----------
__global__ __launch_bounds__(256)
void k_gnstat(const __bf16* __restrict__ wraw, float* __restrict__ mu,
              float* __restrict__ rs)
{
    __shared__ float ssum[256], ssq[256];
    int g = blockIdx.x;                 // b*32 + cg
    int b = g >> 5, cg = g & 31;
    const __bf16* p = wraw + (size_t)b * WCHC * SP + (size_t)cg * 9 * SP;
    float s = 0.f, q = 0.f;
    for (int i = threadIdx.x; i < 9 * SP; i += 256) {
        float v = (float)p[i];
        s += v; q += v * v;
    }
    ssum[threadIdx.x] = s; ssq[threadIdx.x] = q;
    __syncthreads();
    for (int st = 128; st > 0; st >>= 1) {
        if (threadIdx.x < st) {
            ssum[threadIdx.x] += ssum[threadIdx.x + st];
            ssq[threadIdx.x]  += ssq[threadIdx.x + st];
        }
        __syncthreads();
    }
    if (threadIdx.x == 0) {
        float inv  = 1.f / (9.f * SP);
        float mean = ssum[0] * inv;
        float var  = ssq[0] * inv - mean * mean;
        mu[g] = mean;
        rs[g] = rsqrtf(var + EPSF);
    }
}

// ---------------- SKA (spatially-adaptive 3x3 dw) + out BN + residual ---------
__global__ __launch_bounds__(256)
void k_ska(const float* __restrict__ x, const __bf16* __restrict__ wraw,
           const float* __restrict__ mu, const float* __restrict__ rs,
           const float* __restrict__ gn_g, const float* __restrict__ gn_b,
           const float* __restrict__ prm, float* __restrict__ out)
{
    int gid   = blockIdx.x;             // 16*256*25
    int chunk = gid % 25;
    int c     = (gid / 25) & 255;
    int b     = gid / (25 * 256);
    int sp    = chunk * 256 + threadIdx.x;
    int h = sp / WD, w = sp % WD;
    int cg = c & 31;
    const float*  xb = x + ((size_t)b * DIMC + c) * SP;
    const __bf16* wp = wraw + ((size_t)b * WCHC + cg * 9) * SP + sp;
    float mean = mu[b * 32 + cg], rsig = rs[b * 32 + cg];
    float acc = 0.f;
#pragma unroll
    for (int i = 0; i < 3; ++i) {
        int hh = h + i - 1;
#pragma unroll
        for (int j = 0; j < 3; ++j) {
            int ww = w + j - 1;
            int k  = i * 3 + j;
            float wn = ((float)wp[(size_t)k * SP] - mean) * rsig * gn_g[cg * 9 + k]
                       + gn_b[cg * 9 + k];
            float xv = 0.f;
            if ((unsigned)hh < HD && (unsigned)ww < WD) xv = xb[hh * WD + ww];
            acc += xv * wn;
        }
    }
    out[((size_t)b * DIMC + c) * SP + sp] = acc * prm[640 + c] + prm[896 + c] + xb[sp];
}

// ---------------- launch ------------------------------------------------------
extern "C" void kernel_launch(void* const* d_in, const int* in_sizes, int n_in,
                              void* d_out, int out_size, void* d_ws, size_t ws_size,
                              hipStream_t stream)
{
    const float* x     = (const float*)d_in[0];
    const float* cv1_w = (const float*)d_in[1];
    const float* cv1_g = (const float*)d_in[2];
    const float* cv1_b = (const float*)d_in[3];
    const float* cv1_m = (const float*)d_in[4];
    const float* cv1_v = (const float*)d_in[5];
    const float* cv2_w = (const float*)d_in[6];
    const float* cv2_g = (const float*)d_in[7];
    const float* cv2_b = (const float*)d_in[8];
    const float* cv2_m = (const float*)d_in[9];
    const float* cv2_v = (const float*)d_in[10];
    const float* cv3_w = (const float*)d_in[11];
    const float* cv3_g = (const float*)d_in[12];
    const float* cv3_b = (const float*)d_in[13];
    const float* cv3_m = (const float*)d_in[14];
    const float* cv3_v = (const float*)d_in[15];
    const float* cv4_w = (const float*)d_in[16];
    const float* cv4_b = (const float*)d_in[17];
    const float* gn_g  = (const float*)d_in[18];
    const float* gn_b  = (const float*)d_in[19];
    const float* bn_g  = (const float*)d_in[20];
    const float* bn_b  = (const float*)d_in[21];
    const float* bn_m  = (const float*)d_in[22];
    const float* bn_v  = (const float*)d_in[23];

    char* ws = (char*)d_ws;
    __bf16* w1  = (__bf16*)(ws + 0);                    //  64 KB
    __bf16* w3  = (__bf16*)(ws + 65536);                //  32 KB
    __bf16* w4  = (__bf16*)(ws + 98304);                //  72 KB
    float*  wdw = (float*) (ws + 172032);               //  25 KB
    float*  prm = (float*) (ws + 197120);               // 4.6 KB
    float*  mu  = (float*) (ws + 201728);               //  2 KB
    float*  rs  = (float*) (ws + 203776);               //  2 KB
    __bf16* h1  = (__bf16*)(ws + 205824);               // 26.2 MB
    __bf16* h2  = (__bf16*)(ws + 205824 + 26214400);    // 26.2 MB
    __bf16* wr  = (__bf16*)(ws + 205824 + 2 * 26214400);// 59.0 MB

    k_prep<<<144, 256, 0, stream>>>(cv1_w, cv1_g, cv1_b, cv1_m, cv1_v,
                                    cv2_w, cv2_g, cv2_b, cv2_m, cv2_v,
                                    cv3_w, cv3_g, cv3_b, cv3_m, cv3_v,
                                    cv4_w, bn_g, bn_b, bn_m, bn_v,
                                    w1, w3, w4, wdw, prm);
    k_cv1   <<<800,    256, 0, stream>>>(x, w1, prm, h1);
    k_dw7   <<<51200,  256, 0, stream>>>(h1, wdw, prm, h2);
    k_cv34  <<<800,    256, 0, stream>>>(h2, w3, w4, cv4_b, prm, wr);
    k_gnstat<<<512,    256, 0, stream>>>(wr, mu, rs);
    k_ska   <<<102400, 256, 0, stream>>>(x, wr, mu, rs, gn_g, gn_b, prm,
                                         (float*)d_out);
}
